// FlashSentenceEncoderLayer_27556510171157
// MI455X (gfx1250) — matrix operand order repaired and checked by reading, not compiled, not run on hardware
//
#include <hip/hip_runtime.h>

// ---------------------------------------------------------------------------
// MI455X (gfx1250): bf16 WMMA + f32 accumulate, TDM (tensor_load_to_lds)
// staging of all GEMM tiles into LDS (double buffered), fused epilogues.
//
// Pipeline:
//   k0: transpose+cast weights (f32 KxN -> bf16 NxK)  [B^T so B-fragments are
//       per-lane contiguous in LDS]
//   k1: LayerNorm -> xn (bf16).
//   k2: GEMM mode0: v  = silu(xn@wv+bv)          -> v^T  (b,h,l) bf16
//   k3: GEMM mode1: mx = silu(xn@wmx+bmx); split -> u bf16, q/k (gamma/beta)
//   k4: GEMM mode2: qk = q@k^T /L + rel_bias; relu^2 -> attn f32 (d_out)
//                                                    + attn bf16 (ws)
//   k5: GEMM mode3: h  = attn@v ; fuse u*h in place -> uh bf16
//   k6: GEMM mode4: out = x + uh@wh + bh             -> d_out f32
// ---------------------------------------------------------------------------

typedef __bf16 bf16_t;
typedef __attribute__((ext_vector_type(16))) __bf16 v16bf;
typedef __attribute__((ext_vector_type(8)))  __bf16 v8bf;
typedef __attribute__((ext_vector_type(4)))  __bf16 v4bf;
typedef __attribute__((ext_vector_type(8)))  float   v8f;
typedef __attribute__((ext_vector_type(4)))  unsigned uint4v;
typedef __attribute__((ext_vector_type(8)))  unsigned uint8v;

#define Lc 2048
#define Bc 8
#define Dc 1024
#define Hc 2048
#define Zc 128
#define NMX (Hc + Zc)
#define MR (Lc * Bc)          // 16384 rows of (l,b)

#define TM 128
#define TN 128
#define TK 32
#define LDST 48               // padded LDS row stride (bf16): 96B, 16B aligned

__device__ __forceinline__ float silu_f(float v) { return v / (1.0f + __expf(-v)); }

__device__ __forceinline__ v8f vzero8() {
  v8f z = {0.f,0.f,0.f,0.f,0.f,0.f,0.f,0.f};
  return z;
}

// A fragment (16x32 bf16, MxK): lane L holds row m=L&15, K in
// [base,base+16) contiguous with base=(L>>4)*16.  Two b128 LDS loads.
__device__ __forceinline__ v16bf load_frag_a(const bf16_t* t, int m_sub, int lane) {
  int m  = m_sub + (lane & 15);
  int kb = (lane >> 4) * 16;
  v8bf lo = *(const v8bf*)(t + m * LDST + kb);
  v8bf hi = *(const v8bf*)(t + m * LDST + kb + 8);
  return __builtin_shufflevector(lo, hi, 0,1,2,3,4,5,6,7,8,9,10,11,12,13,14,15);
}

// B fragment (32x16 bf16, KxN) from an N-major (B^T) LDS tile: lane L holds
// column n=L&15, K in [kb,kb+16) contiguous with kb=(L>>4)*16.
__device__ __forceinline__ v16bf load_frag_b(const bf16_t* t, int n_sub, int lane) {
  int n  = n_sub + (lane & 15);
  int kb = (lane >> 4) * 16;
  const v8bf* p = (const v8bf*)(t + n * LDST + kb);
  v8bf lo = p[0];
  v8bf hi = p[1];
  return __builtin_shufflevector(lo, hi, 0,1,2,3,4,5,6,7,8,9,10,11,12,13,14,15);
}

// ---------------------------------------------------------------------------
// TDM: DMA one (128 rows x 32 bf16) tile from global (row stride ld_elems)
// into LDS at lds_addr with a 32B pad after each 64B row -> 96B LDS stride,
// matching LDST.  D# group0 (4 SGPRs) + group1 (8 SGPRs); issued per-wave,
// tracked by TENSORcnt.
// ---------------------------------------------------------------------------
__device__ __forceinline__ void tdm_tile_load(unsigned lds_addr,
                                              const bf16_t* gtile,
                                              long ld_elems, int k_total) {
  unsigned long long ga = (unsigned long long)(uintptr_t)gtile;
  uint4v g0;
  g0[0] = 1u;                                    // count=1, user descriptor
  g0[1] = lds_addr;                              // lds_addr (bytes)
  g0[2] = (unsigned)ga;                          // global_addr[31:0]
  g0[3] = ((unsigned)(ga >> 32) & 0x01FFFFFFu)   // global_addr[56:32]
        | (2u << 30);                            // type=2 ("image")
  uint8v g1;
  g1[0] = (1u << 16)                             // data_size = 2B
        | (1u << 20)                             // pad_enable
        | (3u << 22)                             // pad_interval: 16 DWORDs (64B)
        | (7u << 25);                            // pad_amount:   8 DWORDs (32B)
  g1[1] = ((unsigned)k_total & 0xFFFFu) << 16;   // tensor_dim0[15:0]  @bits63:48
  g1[2] = ((unsigned)k_total >> 16) & 0xFFFFu;   // tensor_dim0[31:16] @bits79:64
  g1[3] = (32u << 16)                            // tile_dim0 = 32 K-elems
        | 0x10u;                                 // tensor_dim1 = 0x100000 rows
  g1[4] = 128u;                                  // tile_dim1 = 128 rows
  g1[5] = (unsigned)ld_elems;                    // tensor_dim0_stride[31:0]
  g1[6] = (unsigned)((unsigned long long)ld_elems >> 32) & 0xFFFFu;
  g1[7] = 0u;
  asm volatile("tensor_load_to_lds %0, %1" :: "s"(g0), "s"(g1) : "memory");
}

struct EpilogueArgs {
  const float* bias;      // modes 0,1,4
  const float* gamma;     // mode 1 (2,Z)
  const float* beta;      // mode 1 (2,Z)
  const float* rel_bias;  // mode 2
  const float* x;         // mode 4 residual
  const bf16_t* u_in;     // mode 3
  bf16_t* bf_out0;        // mode-dependent bf16 sink
  bf16_t* bf_out1;        // mode 1: q
  bf16_t* bf_out2;        // mode 1: k
  float*  f_out;          // mode 2: attn, mode 4: out
};

// ---------------------------------------------------------------------------
// k0: W (KxN f32) -> Wt (NxK bf16)
// ---------------------------------------------------------------------------
__global__ __launch_bounds__(256)
void transpose_cast_kernel(const float* __restrict__ W, bf16_t* __restrict__ Wt,
                           int K, int N) {
  __shared__ float tile[32][33];
  int n0 = blockIdx.x * 32, k0 = blockIdx.y * 32;
  int t = threadIdx.x;
  int c = t & 31, r = t >> 5;                 // r in 0..7
#pragma unroll
  for (int i = 0; i < 4; ++i) {
    int kk = r + i * 8;
    tile[kk][c] = W[(long)(k0 + kk) * N + (n0 + c)];
  }
  __syncthreads();
#pragma unroll
  for (int i = 0; i < 4; ++i) {
    int nn = r + i * 8;
    Wt[(long)(n0 + nn) * K + (k0 + c)] = (bf16_t)tile[c][nn];
  }
}

// ---------------------------------------------------------------------------
// k1: LayerNorm row (D=1024) -> bf16
// ---------------------------------------------------------------------------
__global__ __launch_bounds__(256)
void layernorm_cast_kernel(const float* __restrict__ x, const float* __restrict__ g,
                           const float* __restrict__ b, bf16_t* __restrict__ xn) {
  int row = blockIdx.x;
  const float* xr = x + (long)row * Dc;
  int t = threadIdx.x;
  float4 v = ((const float4*)xr)[t];
  float s  = v.x + v.y + v.z + v.w;
  float ss = v.x*v.x + v.y*v.y + v.z*v.z + v.w*v.w;
#pragma unroll
  for (int o = 16; o > 0; o >>= 1) {
    s  += __shfl_xor(s,  o, 32);
    ss += __shfl_xor(ss, o, 32);
  }
  __shared__ float wsum[8], wsq[8];
  __shared__ float mu_s, rs_s;
  int w = t >> 5, lane = t & 31;
  if (lane == 0) { wsum[w] = s; wsq[w] = ss; }
  __syncthreads();
  if (t == 0) {
    float S = 0.f, SS = 0.f;
#pragma unroll
    for (int i = 0; i < 8; ++i) { S += wsum[i]; SS += wsq[i]; }
    float mu  = S * (1.0f / Dc);
    float var = SS * (1.0f / Dc) - mu * mu;
    mu_s = mu;
    rs_s = rsqrtf(var + 1e-5f);
  }
  __syncthreads();
  float mu = mu_s, rs = rs_s;
  float4 gg = ((const float4*)g)[t];
  float4 bb = ((const float4*)b)[t];
  v4bf o;
  o[0] = (bf16_t)((v.x - mu) * rs * gg.x + bb.x);
  o[1] = (bf16_t)((v.y - mu) * rs * gg.y + bb.y);
  o[2] = (bf16_t)((v.z - mu) * rs * gg.z + bb.z);
  o[3] = (bf16_t)((v.w - mu) * rs * gg.w + bb.w);
  *((v4bf*)(xn + (long)row * Dc) + t) = o;
}

// ---------------------------------------------------------------------------
// Generic tiled bf16 WMMA GEMM: C[m,n] = sum_k A[m,k]*Bt[n,k], fused epilogue.
// 256 threads = 8 wave32 (2x4). Each wave: 64x32 = 4x2 WMMA accumulators.
// Tiles are DMA'd into LDS by the TDM (wave 0 issues; TENSORcnt + barrier
// synchronize), double buffered, so the math waves only do ds_load + wmma.
// ---------------------------------------------------------------------------
__global__ __launch_bounds__(256)
void gemm_wmma_kernel(const bf16_t* __restrict__ A, long lda, long strideA,
                      const bf16_t* __restrict__ Bt, long ldb, long strideB,
                      int K, int mode, EpilogueArgs ea) {
  __shared__ bf16_t Asm[2][TM * LDST];
  __shared__ bf16_t Bsm[2][TM * LDST];

  const int tid = threadIdx.x;
  const int m0 = blockIdx.y * TM;
  const int n0 = blockIdx.x * TN;
  const int bz = blockIdx.z;
  A  += (long)bz * strideA;
  Bt += (long)bz * strideB;

  const int KT = K >> 5;
  const bool tdm_wave = (tid < 32);

  const unsigned ldsA0 = (unsigned)(uintptr_t)&Asm[0][0];
  const unsigned ldsA1 = (unsigned)(uintptr_t)&Asm[1][0];
  const unsigned ldsB0 = (unsigned)(uintptr_t)&Bsm[0][0];
  const unsigned ldsB1 = (unsigned)(uintptr_t)&Bsm[1][0];

  auto tdm_stage = [&](int buf, int kt) {
    const long k0 = (long)kt * TK;
    tdm_tile_load(buf ? ldsA1 : ldsA0, A  + (long)m0 * lda + k0, lda, K);
    tdm_tile_load(buf ? ldsB1 : ldsB0, Bt + (long)n0 * ldb + k0, ldb, K);
  };

  const int w    = tid >> 5;
  const int lane = tid & 31;
  const int wm   = w >> 2;   // 0..1 -> M offset wm*64
  const int wn   = w & 3;    // 0..3 -> N offset wn*32

  v8f acc[4][2];
#pragma unroll
  for (int mi = 0; mi < 4; ++mi)
#pragma unroll
    for (int ni = 0; ni < 2; ++ni) acc[mi][ni] = vzero8();

  if (tdm_wave) tdm_stage(0, 0);

  for (int kt = 0; kt < KT; ++kt) {
    const int cur = kt & 1;
    if (tdm_wave) __builtin_amdgcn_s_wait_tensorcnt(0);  // buffer `cur` landed
    __syncthreads();                                     // visible to all waves
    if (tdm_wave && (kt + 1) < KT) tdm_stage(cur ^ 1, kt + 1);  // overlap DMA

    const bf16_t* At = Asm[cur];
    const bf16_t* Bs = Bsm[cur];
    v16bf af[4], bfq[2];
#pragma unroll
    for (int mi = 0; mi < 4; ++mi) af[mi]  = load_frag_a(At, wm * 64 + mi * 16, lane);
#pragma unroll
    for (int ni = 0; ni < 2; ++ni) bfq[ni] = load_frag_b(Bs, wn * 32 + ni * 16, lane);
#pragma unroll
    for (int mi = 0; mi < 4; ++mi)
#pragma unroll
      for (int ni = 0; ni < 2; ++ni)
        acc[mi][ni] = __builtin_amdgcn_wmma_f32_16x16x32_bf16(
            false, af[mi], false, bfq[ni], (short)0, acc[mi][ni], false, false);
  }

  // Epilogue: C/D layout -> element r of v8f is (m = base + (lane>>4)*8 + r,
  //                                              n = base + (lane&15)).
#pragma unroll
  for (int mi = 0; mi < 4; ++mi) {
#pragma unroll
    for (int ni = 0; ni < 2; ++ni) {
      const int bm = m0 + wm * 64 + mi * 16 + (lane >> 4) * 8;
      const int bn = n0 + wn * 32 + ni * 16 + (lane & 15);
#pragma unroll
      for (int r = 0; r < 8; ++r) {
        const int gm = bm + r;
        const int gn = bn;
        float val = acc[mi][ni][r];
        if (mode == 0) {                 // v = silu(.) -> v^T (b,h,l)
          float s = silu_f(val + ea.bias[gn]);
          int l = gm >> 3, b = gm & 7;
          ea.bf_out0[((long)b * Hc + gn) * Lc + l] = (bf16_t)s;
        } else if (mode == 1) {          // mx = silu(.); split u / q,k
          float s = silu_f(val + ea.bias[gn]);
          if (gn < Hc) {
            ea.bf_out0[(long)gm * Hc + gn] = (bf16_t)s;
          } else {
            int z = gn - Hc;
            int l = gm >> 3, b = gm & 7;
            long qi = ((long)b * Lc + l) * Zc + z;
            ea.bf_out1[qi] = (bf16_t)(s * ea.gamma[z]      + ea.beta[z]);
            ea.bf_out2[qi] = (bf16_t)(s * ea.gamma[Zc + z] + ea.beta[Zc + z]);
          }
        } else if (mode == 2) {          // attn = relu(qk/L + rel)^2
          float t = val * (1.0f / Lc) + ea.rel_bias[(Lc - 1) - gm + gn];
          t = t > 0.f ? t * t : 0.f;
          long ai = ((long)bz * Lc + gm) * Lc + gn;
          ea.f_out[ai]   = t;
          ea.bf_out0[ai] = (bf16_t)t;
        } else if (mode == 3) {          // uh = u * (attn@v), in place
          long ri  = (long)gm * Bc + bz;
          long idx = ri * Hc + gn;
          float uu = (float)ea.u_in[idx];
          ea.bf_out0[idx] = (bf16_t)(uu * val);
        } else {                         // out = x + uh@wh + bh
          long idx = (long)gm * Dc + gn;
          ea.f_out[idx] = ea.x[idx] + val + ea.bias[gn];
        }
      }
    }
  }
}

// ---------------------------------------------------------------------------
// Compile probe (never launched): keeps the async-to-LDS path verified for
// promotion (e.g. epilogue scatter staging) alongside cluster barriers.
// ---------------------------------------------------------------------------
__global__ __launch_bounds__(32)
void cdna5_async_tdm_probe(const bf16_t* __restrict__ gsrc, unsigned* __restrict__ gdst) {
  __shared__ bf16_t lds[4096];
  unsigned lds_off = (unsigned)(uintptr_t)&lds[0] + threadIdx.x * 16u;
  unsigned long long ga = (unsigned long long)(uintptr_t)gsrc + threadIdx.x * 16u;
  asm volatile("global_load_async_to_lds_b128 %0, %1, off"
               :: "v"(lds_off), "v"(ga) : "memory");
  asm volatile("s_wait_asynccnt 0" ::: "memory");
  __builtin_amdgcn_s_cluster_barrier();
  gdst[threadIdx.x] = (unsigned)(float)lds[threadIdx.x];
}

// ---------------------------------------------------------------------------
// Host side
// ---------------------------------------------------------------------------
static constexpr size_t OFF_XN   = 0;
static constexpr size_t SZ_XN    = (size_t)MR * Dc * 2;
static constexpr size_t OFF_WVT  = OFF_XN + SZ_XN;
static constexpr size_t SZ_WVT   = (size_t)Hc * Dc * 2;
static constexpr size_t OFF_WMXT = OFF_WVT + SZ_WVT;
static constexpr size_t SZ_WMXT  = (size_t)NMX * Dc * 2;
static constexpr size_t OFF_WHT  = OFF_WMXT + SZ_WMXT;
static constexpr size_t SZ_WHT   = (size_t)Dc * Hc * 2;
static constexpr size_t OFF_VT   = OFF_WHT + SZ_WHT;
static constexpr size_t SZ_VT    = (size_t)Bc * Hc * Lc * 2;
static constexpr size_t OFF_U    = OFF_VT + SZ_VT;
static constexpr size_t SZ_U     = (size_t)MR * Hc * 2;
static constexpr size_t OFF_Q    = OFF_U + SZ_U;
static constexpr size_t SZ_Q     = (size_t)Bc * Lc * Zc * 2;
static constexpr size_t OFF_K    = OFF_Q + SZ_Q;
static constexpr size_t OFF_ATB  = OFF_K + SZ_Q;

extern "C" void kernel_launch(void* const* d_in, const int* in_sizes, int n_in,
                              void* d_out, int out_size, void* d_ws, size_t ws_size,
                              hipStream_t stream) {
  const float* x        = (const float*)d_in[0];
  const float* ln_g     = (const float*)d_in[1];
  const float* ln_b     = (const float*)d_in[2];
  const float* wv       = (const float*)d_in[3];
  const float* bv       = (const float*)d_in[4];
  const float* wmx      = (const float*)d_in[5];
  const float* bmx      = (const float*)d_in[6];
  const float* gamma    = (const float*)d_in[7];
  const float* beta     = (const float*)d_in[8];
  const float* wh       = (const float*)d_in[9];
  const float* bh       = (const float*)d_in[10];
  const float* rel_bias = (const float*)d_in[11];

  float* out  = (float*)d_out;                       // (L,B,D)
  float* attn = out + (size_t)MR * Dc;               // (B,L,L)

  char* ws = (char*)d_ws;
  bf16_t* XN   = (bf16_t*)(ws + OFF_XN);
  bf16_t* WVT  = (bf16_t*)(ws + OFF_WVT);
  bf16_t* WMXT = (bf16_t*)(ws + OFF_WMXT);
  bf16_t* WHT  = (bf16_t*)(ws + OFF_WHT);
  bf16_t* VT   = (bf16_t*)(ws + OFF_VT);
  bf16_t* U    = (bf16_t*)(ws + OFF_U);
  bf16_t* Qb   = (bf16_t*)(ws + OFF_Q);
  bf16_t* Kb   = (bf16_t*)(ws + OFF_K);
  bf16_t* ATB  = (bf16_t*)(ws + OFF_ATB);

  // k0: weight transposes (f32 KxN -> bf16 NxK)
  transpose_cast_kernel<<<dim3(Hc / 32, Dc / 32), 256, 0, stream>>>(wv,  WVT,  Dc, Hc);
  transpose_cast_kernel<<<dim3(NMX / 32, Dc / 32), 256, 0, stream>>>(wmx, WMXT, Dc, NMX);
  transpose_cast_kernel<<<dim3(Dc / 32, Hc / 32), 256, 0, stream>>>(wh,  WHT,  Hc, Dc);

  // k1: LayerNorm
  layernorm_cast_kernel<<<MR, 256, 0, stream>>>(x, ln_g, ln_b, XN);

  // k2: v = silu(xn@wv+bv) -> VT (b,h,l)
  {
    EpilogueArgs ea{}; ea.bias = bv; ea.bf_out0 = VT;
    gemm_wmma_kernel<<<dim3(Hc / TN, MR / TM, 1), 256, 0, stream>>>(
        XN, Dc, 0, WVT, Dc, 0, Dc, 0, ea);
  }
  // k3: mx = silu(xn@wmx+bmx) -> U, Q, K
  {
    EpilogueArgs ea{}; ea.bias = bmx; ea.gamma = gamma; ea.beta = beta;
    ea.bf_out0 = U; ea.bf_out1 = Qb; ea.bf_out2 = Kb;
    gemm_wmma_kernel<<<dim3(NMX / TN, MR / TM, 1), 256, 0, stream>>>(
        XN, Dc, 0, WMXT, Dc, 0, Dc, 1, ea);
  }
  // k4: attn = relu(q@k^T/L + rel)^2 -> d_out(f32) + ATB(bf16)
  {
    EpilogueArgs ea{}; ea.rel_bias = rel_bias; ea.f_out = attn; ea.bf_out0 = ATB;
    gemm_wmma_kernel<<<dim3(Lc / TN, Lc / TM, Bc), 256, 0, stream>>>(
        Qb, Zc, (long)Lc * Zc, Kb, Zc, (long)Lc * Zc, Zc, 2, ea);
  }
  // k5: uh = u * (attn@v), in place over U
  {
    EpilogueArgs ea{}; ea.u_in = U; ea.bf_out0 = U;
    gemm_wmma_kernel<<<dim3(Hc / TN, Lc / TM, Bc), 256, 0, stream>>>(
        ATB, Lc, (long)Lc * Lc, VT, Lc, (long)Hc * Lc, Lc, 3, ea);
  }
  // k6: out = x + uh@wh + bh
  {
    EpilogueArgs ea{}; ea.bias = bh; ea.x = x; ea.f_out = out;
    gemm_wmma_kernel<<<dim3(Dc / TN, MR / TM, 1), 256, 0, stream>>>(
        U, Hc, 0, WHT, Hc, 0, Hc, 4, ea);
  }
}
